// LIFLinearLayer_54683523612726
// MI455X (gfx1250) — matrix operand order repaired and checked by reading, not compiled, not run on hardware
//
#include <hip/hip_runtime.h>

// ---------------------------------------------------------------------------
// LIF linear layer for MI455X (gfx1250, wave32, WMMA, async-to-LDS DMA).
//   Kernel 1: currents = x @ W^T + b.
//     - block tile 128(M) x 128(N), 8 waves (2x4), wave tile 64x32 (4x2 WMMA)
//     - A and W panels staged into double-buffered LDS with
//       global_load_async_to_lds_b128 (ASYNCcnt) overlapping the WMMAs
//     - f32 -> f16 convert on LDS read, f32 accumulate via
//       v_wmma_f32_16x16x32_f16
//   Kernel 2: sequential LIF scan over T per (b,d) channel, in place.
//   Kernel 3: integer spike count -> float sum (deterministic).
// ---------------------------------------------------------------------------

typedef __attribute__((ext_vector_type(16))) _Float16 v16h;
typedef __attribute__((ext_vector_type(8)))  float    v8f;
typedef __attribute__((ext_vector_type(4)))  float    v4f;

#define B_   128
#define T_   512
#define DIN  512
#define DOUT 512
#define M_   (B_ * T_)          // 65536 GEMM rows

#define BETA      0.95f
#define THRESHOLD 1.0f

#define KSTEP 32                // K per WMMA step
#define BM    128               // block tile M
#define BN    128               // block tile N
#define TM    4                 // wave tile = 64 x 32
#define TN    2
#define LDS_STRIDE 36           // padded floats/row: 16B-aligned rows,
                                // bank-conflict-free (36*r mod 64 hits 0,4,..,60)

// ---- async DMA: global -> LDS, 16 bytes per lane, ASYNCcnt-tracked ---------
__device__ __forceinline__ void async_copy_b128(unsigned int lds_byte,
                                                unsigned long long gbase,
                                                unsigned int goff_byte) {
  asm volatile("global_load_async_to_lds_b128 %0, %1, %2"
               :
               : "v"(lds_byte), "v"(goff_byte), "s"(gbase)
               : "memory");
}

// Stage one K-step of the A panel (BM x KSTEP) and W panel (BN x KSTEP) into
// LDS buffer `buf`. 1024 b128 chunks per panel, 256 threads -> 4+4 issues each.
__device__ __forceinline__ void stage_panels(unsigned int asBase, unsigned int wsBase,
                                             unsigned long long Xb, unsigned long long Wb,
                                             int M0B, int N0B, int k0, int buf, int tid) {
#pragma unroll
  for (int q = 0; q < 4; ++q) {
    const int c   = q * 256 + tid;       // 0..1023
    const int row = c >> 3;              // 0..127
    const int kq  = (c & 7) * 4;         // 0,4,..,28 (floats)
    const unsigned int lofs =
        (unsigned int)(((buf * BM + row) * LDS_STRIDE + kq) * 4);
    const unsigned int ga =
        (unsigned int)((((M0B + row) * DIN) + k0 + kq) * 4);
    async_copy_b128(asBase + lofs, Xb, ga);
    const unsigned int gw =
        (unsigned int)((((N0B + row) * DIN) + k0 + kq) * 4);
    async_copy_b128(wsBase + lofs, Wb, gw);
  }
}

// Load 8 consecutive f32 (16B-aligned, LDS) and place as f16 in d[base..base+7].
__device__ __forceinline__ void cvt8(const float* __restrict__ p, v16h& d, int base) {
  v4f a = *reinterpret_cast<const v4f*>(p);
  v4f b = *reinterpret_cast<const v4f*>(p + 4);
  d[base + 0] = (_Float16)a[0]; d[base + 1] = (_Float16)a[1];
  d[base + 2] = (_Float16)a[2]; d[base + 3] = (_Float16)a[3];
  d[base + 4] = (_Float16)b[0]; d[base + 5] = (_Float16)b[1];
  d[base + 6] = (_Float16)b[2]; d[base + 7] = (_Float16)b[3];
}

// -------------------- Kernel 1: GEMM (currents = x @ W^T + bias) ------------
__global__ __launch_bounds__(256)
void lif_gemm_kernel(const float* __restrict__ X,     // [M_, DIN]
                     const float* __restrict__ W,     // [DOUT, DIN]
                     const float* __restrict__ bias,  // [DOUT]
                     float* __restrict__ C) {         // [M_, DOUT]
  __shared__ __attribute__((aligned(16))) float As[2][BM][LDS_STRIDE]; // 36 KB
  __shared__ __attribute__((aligned(16))) float Ws[2][BN][LDS_STRIDE]; // 36 KB

  const int tid  = threadIdx.x;
  const int lane = tid & 31;
  const int wv   = tid >> 5;            // 0..7
  const int wm   = wv >> 2;             // 0..1  (M within block)
  const int wn   = wv & 3;              // 0..3  (N within block)
  const int M0B  = (blockIdx.x >> 2) * BM;   // 512 M-blocks
  const int N0B  = (blockIdx.x & 3) * BN;    // 4   N-blocks
  const int lr   = lane & 15;
  const int hi   = (lane >> 4) & 1;

  // LDS byte offsets (low 32 bits of the flat shared address = LDS offset)
  const unsigned int asBase = (unsigned int)(uintptr_t)&As[0][0][0];
  const unsigned int wsBase = (unsigned int)(uintptr_t)&Ws[0][0][0];
  const unsigned long long Xb = (unsigned long long)(uintptr_t)X;
  const unsigned long long Wb = (unsigned long long)(uintptr_t)W;

  v8f acc[TM][TN];
#pragma unroll
  for (int mt = 0; mt < TM; ++mt)
#pragma unroll
    for (int nt = 0; nt < TN; ++nt)
      acc[mt][nt] = (v8f)(0.0f);

  // Prologue: stage buffer 0.
  stage_panels(asBase, wsBase, Xb, Wb, M0B, N0B, 0, 0, tid);

  for (int k0 = 0; k0 < DIN; k0 += KSTEP) {
    const int buf = (k0 / KSTEP) & 1;

    // Our 8 outstanding async loads are exactly the current buffer's.
    asm volatile("s_wait_asynccnt 0x0" ::: "memory");
    __syncthreads();  // all waves' DMA done; all prior-buffer reads retired

    // Kick off the next buffer's DMA; it overlaps the WMMAs below.
    if (k0 + KSTEP < DIN)
      stage_panels(asBase, wsBase, Xb, Wb, M0B, N0B, k0 + KSTEP, buf ^ 1, tid);

    // ---- A fragments: 16x32 (MxK) f16 per ISA layout ----
    v16h afrag[TM];
#pragma unroll
    for (int mt = 0; mt < TM; ++mt) {
      const int rl = wm * 64 + mt * 16 + lr;
      const float* pa = &As[buf][rl][hi * 8];
      cvt8(pa,      afrag[mt], 0);
      cvt8(pa + 16, afrag[mt], 8);
    }

    // ---- B fragments: 32x16 (KxN) f16; B[k][n] = W[n][k] ----
    v16h bfrag[TN];
#pragma unroll
    for (int nt = 0; nt < TN; ++nt) {
      const int cl = wn * 32 + nt * 16 + lr;
      const float* pb = &Ws[buf][cl][hi * 8];
      cvt8(pb,      bfrag[nt], 0);
      cvt8(pb + 16, bfrag[nt], 8);
    }

    // ---- 8 WMMAs per K-step ----
#pragma unroll
    for (int mt = 0; mt < TM; ++mt)
#pragma unroll
      for (int nt = 0; nt < TN; ++nt)
        acc[mt][nt] = __builtin_amdgcn_wmma_f32_16x16x32_f16(
            /*neg_a=*/false, afrag[mt],
            /*neg_b=*/false, bfrag[nt],
            /*c_mod=*/(short)0, acc[mt][nt],
            /*reuse_a=*/false, /*reuse_b=*/false);
  }

  // ---- Epilogue: bias add + store ----
  const int M0 = M0B + wm * 64;
  const int N0 = N0B + wn * 32;
#pragma unroll
  for (int nt = 0; nt < TN; ++nt) {
    const int   col = N0 + nt * 16 + lr;
    const float bv  = bias[col];
#pragma unroll
    for (int mt = 0; mt < TM; ++mt) {
#pragma unroll
      for (int j = 0; j < 8; ++j) {
        const int row = M0 + mt * 16 + hi * 8 + j;
        C[(size_t)row * DOUT + col] = acc[mt][nt][j] + bv;
      }
    }
  }
}

// -------------------- Kernel 2: LIF scan (in-place) -------------------------
__global__ __launch_bounds__(256)
void lif_scan_kernel(float* __restrict__ cur_spk,       // [B_,T_,DOUT] in/out
                     unsigned int* __restrict__ count) {
  const int tid = blockIdx.x * blockDim.x + threadIdx.x; // 0 .. B_*DOUT-1
  const int b = tid >> 9;
  const int d = tid & (DOUT - 1);

  float mem = 0.0f;
  unsigned int cnt = 0;
  size_t idx = (size_t)b * T_ * DOUT + d;
#pragma unroll 4
  for (int t = 0; t < T_; ++t) {
    const float c = cur_spk[idx];
    mem = BETA * mem + c;
    const float spk = (mem > THRESHOLD) ? 1.0f : 0.0f;
    cur_spk[idx] = spk;
    mem -= spk * THRESHOLD;
    cnt += (unsigned int)spk;
    idx += (size_t)DOUT;
  }

  // wave32 reduction, one integer atomic per wave (deterministic)
#pragma unroll
  for (int off = 16; off >= 1; off >>= 1)
    cnt += __shfl_xor(cnt, off, 32);
  if ((threadIdx.x & 31) == 0)
    atomicAdd(count, cnt);
}

// -------------------- Kernel 3: zero / finalize -----------------------------
__global__ void lif_zero_kernel(unsigned int* __restrict__ count) {
  *count = 0u;
}

__global__ void lif_finalize_kernel(float* __restrict__ sum_slot) {
  const unsigned int c = *reinterpret_cast<unsigned int*>(sum_slot);
  *sum_slot = (float)c;
}

// ---------------------------------------------------------------------------
extern "C" void kernel_launch(void* const* d_in, const int* in_sizes, int n_in,
                              void* d_out, int out_size, void* d_ws, size_t ws_size,
                              hipStream_t stream) {
  const float* x    = (const float*)d_in[0];   // [128,512,512]
  const float* W    = (const float*)d_in[1];   // [512,512]
  const float* bias = (const float*)d_in[2];   // [512]

  float* out      = (float*)d_out;                 // spikes, then sum
  float* sum_slot = out + (size_t)M_ * DOUT;       // last element

  // 1) GEMM: (M/128)x(N/128) = 512*4 = 2048 blocks of 8 waves.
  lif_gemm_kernel<<<2048, 256, 0, stream>>>(x, W, bias, out);

  // 2) zero the integer spike counter (stored in the sum slot).
  lif_zero_kernel<<<1, 1, 0, stream>>>((unsigned int*)sum_slot);

  // 3) LIF scan: 65536 channels.
  lif_scan_kernel<<<(B_ * DOUT) / 256, 256, 0, stream>>>(
      out, (unsigned int*)sum_slot);

  // 4) convert integer count to float sum, in place.
  lif_finalize_kernel<<<1, 1, 0, stream>>>(sum_slot);
}
